// CAN_Layer_25709674233957
// MI455X (gfx1250) — compile-verified, matrix operand order checked by reading, not run on hardware
//
#include <hip/hip_runtime.h>
#include <hip/hip_bf16.h>

// ---------------------------------------------------------------------------
// CDNA5 (gfx1250) cross-attention (CAN layer), f16 WMMA with f32 accumulate.
// Wave32 everywhere. 16x16x32 f16 WMMA per CDNA5 ISA.
// Projection GEMM stages its shared weight tile in LDS via
// GLOBAL_LOAD_ASYNC_TO_LDS_B128 (ASYNCcnt path, CDNA5 ISA).
// ---------------------------------------------------------------------------

typedef __attribute__((ext_vector_type(16))) _Float16 v16h;
typedef __attribute__((ext_vector_type(8)))  _Float16 v8h;
typedef __attribute__((ext_vector_type(8)))  float    v8f;

union Frag16 { v16h v; v8h h[2]; };

#define D_MODEL 512
#define NBATCH  32
#define LPG     512   // protein groups (2048 / 4)
#define LDG     256   // drug groups (256 / 1)
#define NHEAD   8
#define DHEAD   64

// ---- 16-bit A/B fragment load (ISA 7.12.2 layout) -------------------------
// lane 0-15 : row/col = lane,    K = {k0+0..7,  k0+16..23}
// lane 16-31: row/col = lane-16, K = {k0+8..15, k0+24..31}
// Both halves are contiguous 16-byte chunks -> two global_load_b128 per lane.
__device__ __forceinline__ v16h ld_frag_g(const _Float16* __restrict__ base,
                                          int ld, int k0) {
  const int lane = threadIdx.x & 31;
  const _Float16* p = base + (size_t)(lane & 15) * ld + k0 + ((lane >> 4) << 3);
  Frag16 f;
  f.h[0] = *reinterpret_cast<const v8h*>(p);
  f.h[1] = *reinterpret_cast<const v8h*>(p + 16);
  return f.v;
}

__device__ __forceinline__ v8f wmma_f16(v16h a, v16h b, v8f c) {
  // (neg_a, A, neg_b, B, c_mod, C, reuse_a, reuse_b)
  return __builtin_amdgcn_wmma_f32_16x16x32_f16(false, a, false, b,
                                                (short)0, c, false, false);
}

// ---- CDNA5 async copy: global -> LDS, tracked with ASYNCcnt ---------------
__device__ __forceinline__ void async_g2l_b128(const void* lds_ptr,
                                               const void* gptr) {
  // Low 32 bits of a generic pointer into LDS == LDS byte offset
  // (LDS aperture occupies addr[63:32]).
  unsigned loff = (unsigned)(unsigned long long)lds_ptr;
  asm volatile("global_load_async_to_lds_b128 %0, %1, off"
               :: "v"(loff), "v"(gptr)
               : "memory");
}

__device__ __forceinline__ void wait_async0() {
#if __has_builtin(__builtin_amdgcn_s_wait_asynccnt)
  __builtin_amdgcn_s_wait_asynccnt(0);
#else
  asm volatile("s_wait_asynccnt 0x0" ::: "memory");
#endif
}

// ---------------------------------------------------------------------------
// Prep kernels
// ---------------------------------------------------------------------------
__global__ void k_group_prot(const float* __restrict__ prot,
                             _Float16* __restrict__ pg, int total) {
  int e = blockIdx.x * blockDim.x + threadIdx.x;
  if (e >= total) return;
  int d = e & (D_MODEL - 1);
  int r = e / D_MODEL;            // n*LPG + g
  int n = r / LPG, g = r - n * LPG;
  const float* src = prot + ((size_t)(n * 2048 + g * 4)) * D_MODEL + d;
  float s = src[0] + src[D_MODEL] + src[2 * D_MODEL] + src[3 * D_MODEL];
  pg[e] = (_Float16)(s * 0.25f);
}

__global__ void k_cast_f16(const float* __restrict__ in,
                           _Float16* __restrict__ out, int total) {
  int e = blockIdx.x * blockDim.x + threadIdx.x;
  if (e >= total) return;
  out[e] = (_Float16)in[e];
}

__global__ void k_mask_prot(const int* __restrict__ m,
                            float* __restrict__ wsm,
                            float* __restrict__ outm, int total) {
  int e = blockIdx.x * blockDim.x + threadIdx.x;
  if (e >= total) return;
  int n = e / LPG, g = e - n * LPG;
  const int* p = m + (size_t)n * 2048 + g * 4;
  float v = (p[0] | p[1] | p[2] | p[3]) ? 1.0f : 0.0f;
  wsm[e] = v;
  outm[e] = v;
}

__global__ void k_mask_copy(const int* __restrict__ m,
                            float* __restrict__ wsm,
                            float* __restrict__ outm, int total) {
  int e = blockIdx.x * blockDim.x + threadIdx.x;
  if (e >= total) return;
  float v = m[e] ? 1.0f : 0.0f;
  wsm[e] = v;
  outm[e] = v;
}

// VT[n][h][d][k] = V[(n*LK + k)*512 + h*64 + d]
__global__ void k_transposeV(const _Float16* __restrict__ V,
                             _Float16* __restrict__ VT, int LK, int total) {
  int e = blockIdx.x * blockDim.x + threadIdx.x;
  if (e >= total) return;
  int k = e % LK; int t = e / LK;
  int d = t % DHEAD; t /= DHEAD;
  int h = t % NHEAD; int n = t / NHEAD;
  VT[e] = V[((size_t)(n * LK + k)) * D_MODEL + h * DHEAD + d];
}

// ---------------------------------------------------------------------------
// Projection GEMM: Y[M,512] = X[M,512](f16) * W^T(f16), W row-major [out][in].
// Block = 8 waves -> 128x64 output tile. The 64x32 W k-slice (4 KB) shared by
// all waves is double-buffered in LDS via async global->LDS b128 copies; each
// wave computes a 16x64 strip with 4 accumulators.
// ---------------------------------------------------------------------------
__global__ __launch_bounds__(256) void k_proj(const _Float16* __restrict__ X,
                                              const _Float16* __restrict__ W,
                                              _Float16* __restrict__ Y, int M) {
  __shared__ __align__(16) _Float16 wtile[2][64 * 32];   // [buf][col][k]

  const int tid  = threadIdx.x;
  const int wave = tid >> 5, lane = tid & 31;
  const int bm = blockIdx.x >> 3;          // 128-row tile
  const int bn = blockIdx.x & 7;           // 64-col tile
  const int m0 = bm * 128 + wave * 16;
  const int n0 = bn * 64;

  // One b128 async copy per thread stages the full 64x32 half-word slice.
  const int srow = tid >> 2;               // 0..63  (output column index)
  const int sch  = (tid & 3) * 8;          // 16-byte chunk within the k-slice
  auto stage = [&](int buf, int k0) {
    async_g2l_b128(&wtile[buf][srow * 32 + sch],
                   W + (size_t)(n0 + srow) * D_MODEL + k0 + sch);
  };

  stage(0, 0);

  const int col = lane & 15, koff = (lane >> 4) << 3;
  v8f acc[4] = {};

  for (int k0 = 0; k0 < D_MODEL; k0 += 32) {
    const int buf = (k0 >> 5) & 1;
    wait_async0();            // my async writes into wtile[buf] are done
    __syncthreads();          // everyone's are; prior reads of wtile[buf^1] done
    if (k0 + 32 < D_MODEL) stage(buf ^ 1, k0 + 32);

    if (k0 + 32 < D_MODEL)
      __builtin_prefetch(X + (size_t)m0 * D_MODEL + k0 + 32, 0, 0);
    v16h a = ld_frag_g(X + (size_t)m0 * D_MODEL, D_MODEL, k0);
#pragma unroll
    for (int j = 0; j < 4; ++j) {
      const _Float16* bp = &wtile[buf][(j * 16 + col) * 32 + koff];
      Frag16 fb;
      fb.h[0] = *reinterpret_cast<const v8h*>(bp);
      fb.h[1] = *reinterpret_cast<const v8h*>(bp + 16);
      acc[j] = wmma_f16(a, fb.v, acc[j]);
    }
  }

  // C/D layout: lanes 0-15 N=lane, VGPR r holds M=r (lo half) / M=8+r (hi half)
  const int mhi = (lane >> 4) << 3;
#pragma unroll
  for (int j = 0; j < 4; ++j)
#pragma unroll
    for (int r = 0; r < 8; ++r)
      Y[(size_t)(m0 + mhi + r) * D_MODEL + n0 + j * 16 + col] =
          (_Float16)acc[j][r];
}

// ---------------------------------------------------------------------------
// Fused attention: one wave handles (batch n, head h, 16-query tile).
//   logits[16, LK] -> LDS (f32), masked softmax per row,
//   alpha -> d_out (strided [..,k,h]) and LDS (f16), then alpha * V via WMMA.
// ---------------------------------------------------------------------------
template <int LK>
__global__ __launch_bounds__(32) void k_attn(
    const _Float16* __restrict__ Q,    // [N*LQ, 512]
    const _Float16* __restrict__ K,    // [N*LK, 512]
    const _Float16* __restrict__ VT,   // [N, 8, 64, LK]
    const float* __restrict__ mrow,    // [N*LQ]
    const float* __restrict__ mcol,    // [N*LK]
    float* __restrict__ alpha_out,     // [N, LQ, LK, 8]
    float* __restrict__ out,           // [N, LQ, 512]
    int LQ) {
  __shared__ __align__(16) float    slog[16 * LK];
  __shared__ __align__(16) _Float16 salpha[16 * LK];

  int b = blockIdx.x;
  const int h  = b % NHEAD; b /= NHEAD;
  const int qt = b % (LQ / 16);
  const int n  = b / (LQ / 16);

  const int lane = threadIdx.x & 31;
  const int col = lane & 15, mhi = (lane >> 4) << 3;

  // ---- logits = Q Kt over dh=64 (two k-steps of 32) ----
  const _Float16* qbase = Q + ((size_t)(n * LQ + qt * 16)) * D_MODEL + h * DHEAD;
  v16h a0 = ld_frag_g(qbase, D_MODEL, 0);
  v16h a1 = ld_frag_g(qbase, D_MODEL, 32);

  for (int kt = 0; kt < LK / 16; ++kt) {
    const _Float16* kbase =
        K + ((size_t)(n * LK + kt * 16)) * D_MODEL + h * DHEAD;
    v16h b0 = ld_frag_g(kbase, D_MODEL, 0);
    v16h b1 = ld_frag_g(kbase, D_MODEL, 32);
    v8f acc = {};
    acc = wmma_f16(a0, b0, acc);
    acc = wmma_f16(a1, b1, acc);
#pragma unroll
    for (int r = 0; r < 8; ++r)
      slog[(mhi + r) * LK + kt * 16 + col] = acc[r];
  }
  __syncthreads();

  // ---- masked softmax over keys; lanes 0-15 each own one query row ----
  if (lane < 16) {
    const int l = qt * 16 + lane;
    const bool rm = mrow[(size_t)n * LQ + l] != 0.0f;
    float* row = slog + lane * LK;
    float mx = -3.0e38f;
    for (int k = 0; k < LK; ++k) {
      float lg = row[k];
      if (!(rm && (mcol[(size_t)n * LK + k] != 0.0f))) lg -= 1.0e6f;
      row[k] = lg;
      mx = fmaxf(mx, lg);
    }
    float s = 0.0f;
    for (int k = 0; k < LK; ++k) s += __expf(row[k] - mx);
    const float inv = 1.0f / s;
    for (int k = 0; k < LK; ++k) {
      float a = rm ? __expf(row[k] - mx) * inv : 0.0f;
      alpha_out[(((size_t)(n * LQ + l)) * LK + k) * NHEAD + h] = a;
      salpha[lane * LK + k] = (_Float16)a;
    }
  }
  __syncthreads();

  // ---- out[16,64] = alpha[16,LK] * V[LK,64] ----
  v8f oacc[4] = {};
  for (int k0 = 0; k0 < LK; k0 += 32) {
    const _Float16* ap = &salpha[(size_t)col * LK + k0 + mhi];
    Frag16 fa;
    fa.h[0] = *reinterpret_cast<const v8h*>(ap);
    fa.h[1] = *reinterpret_cast<const v8h*>(ap + 16);
#pragma unroll
    for (int j = 0; j < 4; ++j) {
      const _Float16* vb =
          VT + ((size_t)((n * NHEAD + h) * DHEAD + j * 16 + col)) * LK + k0 + mhi;
      Frag16 fb;
      fb.h[0] = *reinterpret_cast<const v8h*>(vb);
      fb.h[1] = *reinterpret_cast<const v8h*>(vb + 16);
      oacc[j] = wmma_f16(fa.v, fb.v, oacc[j]);
    }
  }
#pragma unroll
  for (int j = 0; j < 4; ++j)
#pragma unroll
    for (int r = 0; r < 8; ++r)
      out[((size_t)(n * LQ + qt * 16 + mhi + r)) * D_MODEL + h * DHEAD +
          j * 16 + col] = oacc[j][r];
}

// ---------------------------------------------------------------------------
// Host launcher
// ---------------------------------------------------------------------------
extern "C" void kernel_launch(void* const* d_in, const int* in_sizes, int n_in,
                              void* d_out, int out_size, void* d_ws,
                              size_t ws_size, hipStream_t stream) {
  (void)in_sizes; (void)n_in; (void)out_size; (void)ws_size;

  const float* protein = (const float*)d_in[0];
  const float* drug    = (const float*)d_in[1];
  const int*   mask_p  = (const int*)d_in[2];
  const int*   mask_d  = (const int*)d_in[3];
  const float* Wsrc[6] = {(const float*)d_in[4], (const float*)d_in[5],
                          (const float*)d_in[6], (const float*)d_in[7],
                          (const float*)d_in[8], (const float*)d_in[9]};

  // ---- workspace carve-out (all f16 unless noted) ----
  char* ws = (char*)d_ws;
  size_t off = 0;
  auto carve = [&](size_t bytes) { char* p = ws + off; off += bytes; return p; };

  const size_t np  = (size_t)NBATCH * LPG * D_MODEL;   // 8,388,608
  const size_t nd  = (size_t)NBATCH * LDG * D_MODEL;   // 4,194,304
  const size_t nw  = (size_t)D_MODEL * D_MODEL;        //   262,144

  _Float16* prot_g = (_Float16*)carve(np * 2);
  _Float16* drug_g = (_Float16*)carve(nd * 2);
  _Float16* Wf[6];
  for (int i = 0; i < 6; ++i) Wf[i] = (_Float16*)carve(nw * 2);
  _Float16* qp = (_Float16*)carve(np * 2);
  _Float16* kp = (_Float16*)carve(np * 2);
  _Float16* vp = (_Float16*)carve(np * 2);
  _Float16* qd = (_Float16*)carve(nd * 2);
  _Float16* kd = (_Float16*)carve(nd * 2);
  _Float16* vd = (_Float16*)carve(nd * 2);
  _Float16* vdT = (_Float16*)carve(nd * 2);            // [N,8,64,256]
  _Float16* vpT = (_Float16*)carve(np * 2);            // [N,8,64,512]
  float* mpf = (float*)carve((size_t)NBATCH * LPG * 4);
  float* mdf = (float*)carve((size_t)NBATCH * LDG * 4);

  // ---- d_out layout (reference tuple order, flat) ----
  float* outp = (float*)d_out;
  float* o_prot = outp;                                   // 32*512*512
  float* o_drug = o_prot + np;                            // 32*256*512
  float* o_mp   = o_drug + nd;                            // 32*512
  float* o_md   = o_mp + (size_t)NBATCH * LPG;            // 32*256
  float* o_apd  = o_md + (size_t)NBATCH * LDG;            // 32*512*256*8
  float* o_adp  = o_apd + (size_t)NBATCH * LPG * LDG * NHEAD;

  const int T = 256;
  // prep
  k_group_prot<<<dim3((np + T - 1) / T), dim3(T), 0, stream>>>(protein, prot_g, (int)np);
  k_cast_f16<<<dim3((nd + T - 1) / T), dim3(T), 0, stream>>>(drug, drug_g, (int)nd);
  for (int i = 0; i < 6; ++i)
    k_cast_f16<<<dim3((nw + T - 1) / T), dim3(T), 0, stream>>>(Wsrc[i], Wf[i], (int)nw);
  k_mask_prot<<<dim3((NBATCH * LPG + T - 1) / T), dim3(T), 0, stream>>>(mask_p, mpf, o_mp, NBATCH * LPG);
  k_mask_copy<<<dim3((NBATCH * LDG + T - 1) / T), dim3(T), 0, stream>>>(mask_d, mdf, o_md, NBATCH * LDG);

  // projections: block = 8 waves = 128x64 tile; grid = (M/128) * 8 col-tiles
  const int Mp = NBATCH * LPG;     // 16384
  const int Md = NBATCH * LDG;     //  8192
  const int blkP = (Mp / 128) * 8; // 1024 blocks
  const int blkD = (Md / 128) * 8; //  512 blocks
  k_proj<<<dim3(blkP), dim3(T), 0, stream>>>(prot_g, Wf[0], qp, Mp);
  k_proj<<<dim3(blkP), dim3(T), 0, stream>>>(prot_g, Wf[1], kp, Mp);
  k_proj<<<dim3(blkP), dim3(T), 0, stream>>>(prot_g, Wf[2], vp, Mp);
  k_proj<<<dim3(blkD), dim3(T), 0, stream>>>(drug_g, Wf[3], qd, Md);
  k_proj<<<dim3(blkD), dim3(T), 0, stream>>>(drug_g, Wf[4], kd, Md);
  k_proj<<<dim3(blkD), dim3(T), 0, stream>>>(drug_g, Wf[5], vd, Md);

  // V transposes for the alpha*V GEMM
  k_transposeV<<<dim3((nd + T - 1) / T), dim3(T), 0, stream>>>(vd, vdT, LDG, (int)nd);
  k_transposeV<<<dim3((np + T - 1) / T), dim3(T), 0, stream>>>(vp, vpT, LPG, (int)np);

  // fused attention (one wave per (n, head, 16-query tile))
  k_attn<LDG><<<dim3(NBATCH * (LPG / 16) * NHEAD), dim3(32), 0, stream>>>(
      qp, kd, vdT, mpf, mdf, o_apd, o_prot, LPG);
  k_attn<LPG><<<dim3(NBATCH * (LDG / 16) * NHEAD), dim3(32), 0, stream>>>(
      qd, kp, vpT, mdf, mpf, o_adp, o_drug, LDG);
}
